// VitTransformerBlock_14353780703339
// MI455X (gfx1250) — compile-verified, matrix operand order checked by reading, not run on hardware
//
#include <hip/hip_runtime.h>
#include <math.h>

typedef __attribute__((ext_vector_type(16))) __bf16 v16bf;
typedef __attribute__((ext_vector_type(8)))  __bf16 v8bf;
typedef __attribute__((ext_vector_type(4)))  __bf16 v4bf;
typedef __attribute__((ext_vector_type(8)))  float  v8f;

#define DEV __device__ __forceinline__

DEV __bf16 to_bf(float x) { return (__bf16)x; }

// Load a 16x32 bf16 fragment in the CDNA5 16-bit A-matrix layout:
// lanes 0-15: row = row0+lane,    K = {kofs+0..7, kofs+16..23}
// lanes 16-31: row = row0+lane-16, K = {kofs+8..15, kofs+24..31}
// (B fragments use the identical gather with row = N column.)
DEV v16bf frag_ld(const __bf16* base, int row0, int stride, int kofs) {
  int l  = threadIdx.x & 31;
  int r  = row0 + (l & 15);
  int k0 = kofs + ((l >> 4) << 3);
  const __bf16* p = base + r * stride + k0;
  v8bf lo = *(const v8bf*)(p);
  v8bf hi = *(const v8bf*)(p + 16);
  v16bf f;
#pragma unroll
  for (int i = 0; i < 8; ++i) { f[i] = lo[i]; f[i + 8] = hi[i]; }
  return f;
}

DEV v8f wmma_bf16(v16bf a, v16bf b, v8f c) {
  return __builtin_amdgcn_wmma_f32_16x16x32_bf16(false, a, false, b, (short)0, c,
                                                 false, false);
}

// ---------------------------------------------------------------- cvt f32->bf16
__global__ __launch_bounds__(256) void cvt_bf16_kernel(const float* __restrict__ s,
                                                       __bf16* __restrict__ d, int n) {
  for (int i = blockIdx.x * 256 + threadIdx.x; i < n; i += gridDim.x * 256)
    d[i] = to_bf(s[i]);
}

// ------------------------------------------------- fused LN0 + LN1 (row = 512)
__global__ __launch_bounds__(256) void ln_fused_kernel(
    const float* __restrict__ x,
    const float* __restrict__ w0, const float* __restrict__ b0,
    const float* __restrict__ w1, const float* __restrict__ b1,
    float* __restrict__ x0, __bf16* __restrict__ hb) {
  int lane = threadIdx.x & 31, wid = threadIdx.x >> 5;
  size_t row = (size_t)blockIdx.x * 8 + wid;
  const float4* xr = (const float4*)(x + row * 512);
  float xv[16];
  float s = 0.f, ss = 0.f;
#pragma unroll
  for (int i = 0; i < 4; ++i) {
    float4 t = xr[lane + 32 * i];
    xv[i * 4 + 0] = t.x; xv[i * 4 + 1] = t.y; xv[i * 4 + 2] = t.z; xv[i * 4 + 3] = t.w;
    s += t.x + t.y + t.z + t.w;
    ss += t.x * t.x + t.y * t.y + t.z * t.z + t.w * t.w;
  }
#pragma unroll
  for (int o = 1; o < 32; o <<= 1) { s += __shfl_xor(s, o, 32); ss += __shfl_xor(ss, o, 32); }
  float mu = s * (1.f / 512.f);
  float var = ss * (1.f / 512.f) - mu * mu;
  float rstd = rsqrtf(var + 1e-5f);
  float yv[16];
  float s1 = 0.f, ss1 = 0.f;
  float4* x0r = (float4*)(x0 + row * 512);
#pragma unroll
  for (int i = 0; i < 4; ++i) {
    float4 t;
    float* tp = (float*)&t;
#pragma unroll
    for (int c = 0; c < 4; ++c) {
      int col = (lane + 32 * i) * 4 + c;
      float y = (xv[i * 4 + c] - mu) * rstd * w0[col] + b0[col];
      yv[i * 4 + c] = y; tp[c] = y;
      s1 += y; ss1 += y * y;
    }
    x0r[lane + 32 * i] = t;
  }
#pragma unroll
  for (int o = 1; o < 32; o <<= 1) { s1 += __shfl_xor(s1, o, 32); ss1 += __shfl_xor(ss1, o, 32); }
  float mu1 = s1 * (1.f / 512.f);
  float rstd1 = rsqrtf(ss1 * (1.f / 512.f) - mu1 * mu1 + 1e-5f);
  __bf16* hr = hb + row * 512;
#pragma unroll
  for (int i = 0; i < 4; ++i) {
    v4bf h;
#pragma unroll
    for (int c = 0; c < 4; ++c) {
      int col = (lane + 32 * i) * 4 + c;
      h[c] = to_bf((yv[i * 4 + c] - mu1) * rstd1 * w1[col] + b1[col]);
    }
    *(v4bf*)(hr + (lane + 32 * i) * 4) = h;
  }
}

// ------------------------------------------------- tiled bf16 WMMA GEMM 128x128x32
// EPI 0: QKV scatter to [B,H,P,N,64] bf16 x3
// EPI 1: + bias + resf -> outf (f32) and ob0 (bf16)
// EPI 2: gelu(+bias) -> ob0 (bf16)
// EPI 3: + bias + resf -> outf (f32)
template <int EPI>
__global__ __launch_bounds__(256) void gemm_bf16_kernel(
    const __bf16* __restrict__ A, const __bf16* __restrict__ B,
    int M, int N, int K,
    const float* __restrict__ bias, const float* __restrict__ resf,
    float* __restrict__ outf,
    __bf16* __restrict__ ob0, __bf16* __restrict__ ob1, __bf16* __restrict__ ob2) {
  __shared__ __bf16 As[128 * 40];   // [m][k], stride 40 halfs (80B, 16B aligned)
  __shared__ __bf16 Bs[128 * 40];   // B transposed: [n][k]
  int tid = threadIdx.x, lane = tid & 31, wid = tid >> 5;
  int tilesN = N >> 7;
  int tm = blockIdx.x / tilesN, tn = blockIdx.x % tilesN;
  int m0 = tm << 7, n0 = tn << 7;
  int wm = wid >> 2, wn = wid & 3;  // 2 (M) x 4 (N) waves, wave tile 64x32
  v8f acc[4][2];
#pragma unroll
  for (int i = 0; i < 4; ++i)
#pragma unroll
    for (int j = 0; j < 2; ++j) acc[i][j] = {};
  int nk = K >> 5;
  for (int kt = 0; kt < nk; ++kt) {
    if (kt + 1 < nk)
      __builtin_prefetch((const void*)(A + (size_t)(m0 + (tid >> 1)) * K + ((kt + 1) << 5)), 0, 0);
    __syncthreads();
#pragma unroll
    for (int c2 = 0; c2 < 2; ++c2) {  // A tile: 512 chunks of 8 halfs
      int c = tid + (c2 << 8);
      int row = c >> 2, kc = (c & 3) << 3;
      *(v8bf*)(As + row * 40 + kc) =
          *(const v8bf*)(A + (size_t)(m0 + row) * K + (kt << 5) + kc);
    }
#pragma unroll
    for (int c2 = 0; c2 < 2; ++c2) {  // B tile, transposed into LDS
      int c = tid + (c2 << 8);
      int kr = c >> 4, nc = (c & 15) << 3;
      v8bf bv = *(const v8bf*)(B + (size_t)((kt << 5) + kr) * N + n0 + nc);
#pragma unroll
      for (int i = 0; i < 8; ++i) Bs[(nc + i) * 40 + kr] = bv[i];
    }
    __syncthreads();
    v16bf af[4], bfr[2];
#pragma unroll
    for (int i = 0; i < 4; ++i) af[i] = frag_ld(As, wm * 64 + i * 16, 40, 0);
#pragma unroll
    for (int j = 0; j < 2; ++j) bfr[j] = frag_ld(Bs, wn * 32 + j * 16, 40, 0);
#pragma unroll
    for (int i = 0; i < 4; ++i)
#pragma unroll
      for (int j = 0; j < 2; ++j) acc[i][j] = wmma_bf16(af[i], bfr[j], acc[i][j]);
  }
  // epilogue: C layout => VGPR r holds row r (lanes 0-15) / row r+8 (lanes 16-31)
  int hf = lane >> 4, ln = lane & 15;
#pragma unroll
  for (int i = 0; i < 4; ++i)
#pragma unroll
    for (int j = 0; j < 2; ++j)
#pragma unroll
      for (int r = 0; r < 8; ++r) {
        int m = m0 + wm * 64 + i * 16 + r + hf * 8;
        int n = n0 + wn * 32 + j * 16 + ln;
        float v = acc[i][j][r];
        if (EPI == 0) {
          int which = n >> 9, inner = n & 511;
          int hh = inner >> 6, dd = inner & 63;
          int bb = m >> 13, pp = (m >> 10) & 7, nn = m & 1023;
          size_t idx = ((((size_t)bb * 8 + hh) * 8 + pp) * 1024 + nn) * 64 + dd;
          __bf16* dst = (which == 0) ? ob0 : (which == 1) ? ob1 : ob2;
          dst[idx] = to_bf(v);
        } else if (EPI == 1) {
          v += bias[n] + resf[(size_t)m * N + n];
          outf[(size_t)m * N + n] = v;
          ob0[(size_t)m * N + n] = to_bf(v);
        } else if (EPI == 2) {
          v += bias[n];
          v = 0.5f * v * (1.0f + erff(v * 0.70710678118654752f));
          ob0[(size_t)m * N + n] = to_bf(v);
        } else {
          v += bias[n] + resf[(size_t)m * N + n];
          outf[(size_t)m * N + n] = v;
        }
      }
}

// ------------------------------------------------- flash attention (transposed)
// Computes S^T = K·Q^T and O^T = V^T·P^T so that per-q softmax statistics are
// per-lane scalars (q column = lane&15) and P^T feeds the second WMMA directly
// from registers (C-layout of S^T == B-fragment layout of P^T).
// q,k,v: [BHP=128][N=1024][D=64] bf16. Block: 128 q rows; wave owns 16 q rows.
// Output with buggy rearrange: Ob[(b*8+h)*1024 + n][p*64 + d] (bf16, [16384,512]).
__global__ __launch_bounds__(256) void attn_kernel(
    const __bf16* __restrict__ Q, const __bf16* __restrict__ Kb,
    const __bf16* __restrict__ V, __bf16* __restrict__ Ob) {
  __shared__ __bf16 Qs[128 * 72];      // [qrow][d], stride 72 (144B)
  __shared__ __bf16 Ks[32 * 72];       // [kvrow][d]
  __shared__ __bf16 Vt[64 * 40];       // transposed: [d][kv]
  int tid = threadIdx.x, lane = tid & 31, wid = tid >> 5;
  int qt = blockIdx.x & 7;
  int bhp = blockIdx.x >> 3;  // 0..127
  const __bf16* qp = Q + (size_t)bhp * (1024 * 64);
  const __bf16* kp = Kb + (size_t)bhp * (1024 * 64);
  const __bf16* vp = V + (size_t)bhp * (1024 * 64);
#pragma unroll
  for (int c4 = 0; c4 < 4; ++c4) {  // Q tile 128x64 -> LDS
    int c = tid + (c4 << 8);
    int row = c >> 3, kc = (c & 7) << 3;
    *(v8bf*)(Qs + row * 72 + kc) =
        *(const v8bf*)(qp + (size_t)(qt * 128 + row) * 64 + kc);
  }
  v8f o[4];  // O^T accumulators: tile t covers d = t*16..t*16+15, q = lane&15
#pragma unroll
  for (int t = 0; t < 4; ++t) o[t] = {};
  float m_run = -1e30f, l_run = 0.f;  // per-lane: q column = lane&15
  const float scale = 0.125f;         // 64^-0.5
  int hf = lane >> 4, ln = lane & 15;
  for (int c32 = 0; c32 < 32; ++c32) {
    __syncthreads();
    int kv0 = c32 << 5;
    {  // K chunk 32x64
      int row = tid >> 3, kc = (tid & 7) << 3;
      *(v8bf*)(Ks + row * 72 + kc) =
          *(const v8bf*)(kp + (size_t)(kv0 + row) * 64 + kc);
    }
    {  // V chunk 32x64 -> transposed LDS [d][kv]
      int kvr = tid >> 3, dc = (tid & 7) << 3;
      v8bf vv = *(const v8bf*)(vp + (size_t)(kv0 + kvr) * 64 + dc);
#pragma unroll
      for (int i = 0; i < 8; ++i) Vt[(dc + i) * 40 + kvr] = vv[i];
    }
    __syncthreads();
    // S^T = K (kv x d) @ Q^T (d x q), K-dim = 64
    v16bf bq0 = frag_ld(Qs, wid * 16, 72, 0);
    v16bf bq1 = frag_ld(Qs, wid * 16, 72, 32);
    v8f s[2];  // tile j: kv rows j*16..j*16+15 (per lane: kv = r + hf*8 + j*16)
#pragma unroll
    for (int j = 0; j < 2; ++j) {
      v16bf ak0 = frag_ld(Ks, j * 16, 72, 0);
      v16bf ak1 = frag_ld(Ks, j * 16, 72, 32);
      v8f c = {};
      c = wmma_bf16(ak0, bq0, c);
      c = wmma_bf16(ak1, bq1, c);
#pragma unroll
      for (int r = 0; r < 8; ++r) s[j][r] = c[r] * scale;
    }
    // per-lane softmax over the 16 in-lane kv values + one cross-half combine
    float cm = s[0][0];
#pragma unroll
    for (int r = 1; r < 8; ++r) cm = fmaxf(cm, s[0][r]);
#pragma unroll
    for (int r = 0; r < 8; ++r) cm = fmaxf(cm, s[1][r]);
    cm = fmaxf(cm, __shfl_xor(cm, 16, 32));
    float mnew = fmaxf(m_run, cm);
    float alpha = __expf(m_run - mnew);
    float rs = 0.f;
#pragma unroll
    for (int j = 0; j < 2; ++j)
#pragma unroll
      for (int r = 0; r < 8; ++r) {
        s[j][r] = __expf(s[j][r] - mnew);
        rs += s[j][r];
      }
    rs += __shfl_xor(rs, 16, 32);
    l_run = l_run * alpha + rs;
    m_run = mnew;
#pragma unroll
    for (int t = 0; t < 4; ++t)
#pragma unroll
      for (int r = 0; r < 8; ++r) o[t][r] *= alpha;
    // P^T in C-layout == B-fragment (N=q, K=kv) layout: pack directly.
    v16bf pb;
#pragma unroll
    for (int r = 0; r < 8; ++r) { pb[r] = to_bf(s[0][r]); pb[r + 8] = to_bf(s[1][r]); }
    // O^T += V^T (d x kv) @ P^T (kv x q), K-dim = 32
#pragma unroll
    for (int t = 0; t < 4; ++t) {
      v16bf av = frag_ld(Vt, t * 16, 40, 0);
      o[t] = wmma_bf16(av, pb, o[t]);
    }
  }
  // write out: lane q = ln; element (t, r) is d = t*16 + r + hf*8 (r contiguous)
  int b_ = bhp >> 6, h_ = (bhp >> 3) & 7, p_ = bhp & 7;
  float inv = 1.0f / l_run;
  size_t orow = ((size_t)(b_ * 8 + h_)) * 1024 + (size_t)(qt * 128 + wid * 16 + ln);
  __bf16* op = Ob + orow * 512 + p_ * 64 + hf * 8;
#pragma unroll
  for (int t = 0; t < 4; ++t) {
    v8bf ov;
#pragma unroll
    for (int r = 0; r < 8; ++r) ov[r] = to_bf(o[t][r] * inv);
    *(v8bf*)(op + t * 16) = ov;
  }
}

// ---------------------------------------------------------------- host launch
extern "C" void kernel_launch(void* const* d_in, const int* in_sizes, int n_in,
                              void* d_out, int out_size, void* d_ws, size_t ws_size,
                              hipStream_t stream) {
  const float* x    = (const float*)d_in[0];
  const float* ln0w = (const float*)d_in[1];
  const float* ln0b = (const float*)d_in[2];
  const float* ln1w = (const float*)d_in[3];
  const float* ln1b = (const float*)d_in[4];
  const float* wqkv = (const float*)d_in[5];
  const float* wo   = (const float*)d_in[6];
  const float* bo   = (const float*)d_in[7];
  const float* w1   = (const float*)d_in[8];
  const float* b1   = (const float*)d_in[9];
  const float* w2   = (const float*)d_in[10];
  const float* b2   = (const float*)d_in[11];

  const int M = 16384;  // B*P*N tokens
  char* wp = (char*)d_ws;
  auto take = [&](size_t bytes) {
    char* p = wp;
    wp += (bytes + 255) & ~(size_t)255;
    return p;
  };
  __bf16* wqkvb = (__bf16*)take((size_t)512 * 1536 * 2);
  __bf16* wob   = (__bf16*)take((size_t)512 * 512 * 2);
  __bf16* w1b   = (__bf16*)take((size_t)512 * 2048 * 2);
  __bf16* w2b   = (__bf16*)take((size_t)2048 * 512 * 2);
  float*  x0    = (float*)take((size_t)M * 512 * 4);
  __bf16* hb    = (__bf16*)take((size_t)M * 512 * 2);  // ln1 out; reused for attn out
  __bf16* qb    = (__bf16*)take((size_t)M * 512 * 2);
  __bf16* kb    = (__bf16*)take((size_t)M * 512 * 2);
  __bf16* vb    = (__bf16*)take((size_t)M * 512 * 2);
  float*  r1f   = (float*)take((size_t)M * 512 * 4);
  __bf16* r1b   = (__bf16*)take((size_t)M * 512 * 2);
  __bf16* h2    = (__bf16*)take((size_t)M * 2048 * 2);

  cvt_bf16_kernel<<<1024, 256, 0, stream>>>(wqkv, wqkvb, 512 * 1536);
  cvt_bf16_kernel<<<1024, 256, 0, stream>>>(wo, wob, 512 * 512);
  cvt_bf16_kernel<<<1024, 256, 0, stream>>>(w1, w1b, 512 * 2048);
  cvt_bf16_kernel<<<1024, 256, 0, stream>>>(w2, w2b, 2048 * 512);

  ln_fused_kernel<<<2048, 256, 0, stream>>>(x, ln0w, ln0b, ln1w, ln1b, x0, hb);

  // QKV: [16384,512] @ [512,1536] -> scatter to q/k/v [B,H,P,N,64]
  gemm_bf16_kernel<0><<<128 * 12, 256, 0, stream>>>(hb, wqkvb, M, 1536, 512,
                                                    nullptr, nullptr, nullptr,
                                                    qb, kb, vb);
  // attention -> hb as [B,H,N,512] (buggy rearrange layout)
  attn_kernel<<<1024, 256, 0, stream>>>(qb, kb, vb, hb);
  // Wo: + b_o + x0 residual -> r1f/r1b
  gemm_bf16_kernel<1><<<128 * 4, 256, 0, stream>>>(hb, wob, M, 512, 512,
                                                   bo, x0, r1f, r1b, nullptr, nullptr);
  // MLP1: gelu(r1 @ w1 + b1) -> h2
  gemm_bf16_kernel<2><<<128 * 16, 256, 0, stream>>>(r1b, w1b, M, 2048, 512,
                                                    b1, nullptr, nullptr, h2, nullptr, nullptr);
  // MLP2: h2 @ w2 + b2 + r1 -> d_out (f32)
  gemm_bf16_kernel<3><<<128 * 4, 256, 0, stream>>>(h2, w2b, M, 512, 2048,
                                                   b2, r1f, (float*)d_out,
                                                   nullptr, nullptr, nullptr);
  (void)in_sizes; (void)n_in; (void)out_size; (void)ws_size;
}
